// MultiheadAttention_23201413333385
// MI455X (gfx1250) — matrix operand order repaired and checked by reading, not compiled, not run on hardware
//
#include <hip/hip_runtime.h>
#include <hip/hip_bf16.h>
#include <math.h>

#define D_MODEL 512
#define N_HEADS 8
#define HEAD_DIM 64
#define B_SZ 2
#define T_SEQ 2048
#define M_ROWS (B_SZ * T_SEQ)   /* 4096 */
#define PE_ROWS 4096             /* 2T-1 = 4095, padded to 4096 */

typedef __bf16 v16bf __attribute__((ext_vector_type(16)));
typedef __bf16 v8bf  __attribute__((ext_vector_type(8)));
typedef float  v8f   __attribute__((ext_vector_type(8)));

static __device__ __forceinline__ int lane_id() { return (int)(threadIdx.x & 31u); }

static __device__ __forceinline__ v8f wmma_bf16(v16bf a, v16bf b, v8f c) {
  return __builtin_amdgcn_wmma_f32_16x16x32_bf16(false, a, false, b, (short)0, c, false, false);
}

// ---- WMMA fragment loaders (gfx1250 wave32 layouts, CDNA5 ISA 7.12.2) ----
// A 16x32 bf16: lane m = l&15; lanes<16 hold K {0..7,16..23}, lanes>=16 hold K {8..15,24..31}
static __device__ __forceinline__ v16bf load_a_bf16(const __bf16* A, int ld, int m0, int k0) {
  int l = lane_id();
  const __bf16* p = A + (long)(m0 + (l & 15)) * ld + k0 + ((l & 16) ? 8 : 0);
  v8bf lo = *(const v8bf*)p;
  v8bf hi = *(const v8bf*)(p + 16);
  v16bf r;
#pragma unroll
  for (int i = 0; i < 8; ++i) { r[i] = lo[i]; r[i + 8] = hi[i]; }
  return r;
}
// B 32x16 bf16 from B^T (row n holds K contiguous): lanes<16 K {0..7,16..23}, lanes>=16 K {8..15,24..31}
static __device__ __forceinline__ v16bf load_b_bf16(const __bf16* Bt, long ld, long n0, long k0) {
  int l = lane_id();
  const __bf16* p = Bt + (long)(n0 + (l & 15)) * ld + k0 + ((l & 16) ? 8 : 0);
  v8bf lo = *(const v8bf*)p;
  v8bf hi = *(const v8bf*)(p + 16);
  v16bf r;
#pragma unroll
  for (int i = 0; i < 8; ++i) { r[i] = lo[i]; r[i + 8] = hi[i]; }
  return r;
}

static __device__ __forceinline__ float bperm_f32(int src_lane, float v) {
  int iv = __builtin_bit_cast(int, v);
  int rv = __builtin_amdgcn_ds_bpermute(src_lane << 2, iv);
  return __builtin_bit_cast(float, rv);
}

// ---------------- one-pass fp32 -> bf16 convert (x and the four weight matrices) -------
__global__ void cvt_kernel(const float* __restrict__ in, __bf16* __restrict__ outp, long n8) {
  long i = (long)blockIdx.x * 256 + threadIdx.x;
  if (i >= n8) return;
  v8f a = *(const v8f*)(in + i * 8);
  v8bf b;
#pragma unroll
  for (int j = 0; j < 8; ++j) b[j] = (__bf16)a[j];
  *(v8bf*)(outp + i * 8) = b;
}

// ---------------- sinusoid relative PE (bf16), zero-padded to 4096 rows ----------------
__global__ void pe_kernel(__bf16* __restrict__ pe) {
  long i = (long)blockIdx.x * 256 + threadIdx.x;
  if (i >= (long)PE_ROWS * D_MODEL) return;
  int row = (int)(i / D_MODEL);
  int j = (int)(i % D_MODEL);
  float val = 0.0f;
  if (row < 2 * T_SEQ - 1) {
    float pos = (float)(row - (T_SEQ - 1));
    int half = j >> 1;
    // exp(2*half * (-ln(10000)/D)); -ln(10000)/512 = -0.017988946
    float div = expf((float)(2 * half) * (-0.01798894610046388f));
    float arg = pos * div;
    val = (j & 1) ? cosf(arg) : sinf(arg);
  }
  pe[i] = (__bf16)val;
}

// ---------------- projection GEMM: C = A(Mx512) @ W^T + bias (+add) -------------------
// bf16 operands, fp32 accumulate; each wave owns a 32x64 tile (8 WMMAs / K-step).
// trans=0 -> out[m*512+n];  trans=1 -> out[n*M+m] (val stored transposed for P@V)
__global__ __launch_bounds__(128) void proj_kernel(
    const __bf16* __restrict__ A, const __bf16* __restrict__ W,
    const float* __restrict__ bias, const float* __restrict__ add0,
    const float* __restrict__ add1, __bf16* __restrict__ out0,
    __bf16* __restrict__ out1, int M, int trans) {
  int wg = blockIdx.x * 4 + (threadIdx.x >> 5);
  const int NT = D_MODEL / 64; // 8 column macro-tiles
  int mi = (wg / NT) * 32;
  int ni = (wg % NT) * 64;
  if (mi >= M) return; // wave-uniform

  v8f acc[2][4] = {};
#pragma unroll 2
  for (int k = 0; k < D_MODEL; k += 32) {
    v16bf a0 = load_a_bf16(A, D_MODEL, mi, k);
    v16bf a1 = load_a_bf16(A, D_MODEL, mi + 16, k);
#pragma unroll
    for (int tj = 0; tj < 4; ++tj) {
      v16bf bb = load_b_bf16(W, D_MODEL, ni + tj * 16, k);
      acc[0][tj] = wmma_bf16(a0, bb, acc[0][tj]);
      acc[1][tj] = wmma_bf16(a1, bb, acc[1][tj]);
    }
  }

  int l = lane_id();
  int lo = l & 15;
  int hi = (l >> 4) & 1;
#pragma unroll
  for (int ti = 0; ti < 2; ++ti) {
#pragma unroll
    for (int tj = 0; tj < 4; ++tj) {
      int n = ni + tj * 16 + lo;
      float bval = bias[n];
      float a0v = add0 ? add0[n] : 0.0f;
      float a1v = add1 ? add1[n] : 0.0f;
      if (!trans) {
#pragma unroll
        for (int r = 0; r < 8; ++r) {
          long m = (long)(mi + ti * 16 + hi * 8 + r);
          float c = acc[ti][tj][r] + bval;
          out0[m * D_MODEL + n] = (__bf16)(c + a0v);
          if (out1) out1[m * D_MODEL + n] = (__bf16)(c + a1v);
        }
      } else {
        v8bf pk;
#pragma unroll
        for (int r = 0; r < 8; ++r) pk[r] = (__bf16)(acc[ti][tj][r] + bval + a0v);
        *(v8bf*)(out0 + (long)n * M + mi + ti * 16 + hi * 8) = pk;
      }
    }
  }
}

// ---------------- fused rel-pos flash attention ----------------
// one wave = one (b, h, 16-query tile); keys streamed 32 at a time.
// bd[q,k] = (q+v)·kl[2047+k-q]: computed as a 48-wide banded GEMM (3 G tiles) and
// shifted in-register with ds_bpermute (3 permutes serve both 16-col score tiles).
__global__ __launch_bounds__(128) void attn_kernel(
    const __bf16* __restrict__ qu, const __bf16* __restrict__ qv,
    const __bf16* __restrict__ kc, const __bf16* __restrict__ kl,
    const __bf16* __restrict__ vT, const int* __restrict__ mask,
    float* __restrict__ out) {
  __shared__ __bf16 pst_all[4][16 * 40]; // padded stride 40 to dodge bank conflicts
  int wid = threadIdx.x >> 5;
  __bf16* pst = &pst_all[wid][0];

  int wg = blockIdx.x * 4 + wid;
  int qt = wg & 127;         // T/16 = 128 query tiles
  int h = (wg >> 7) & 7;
  int b = wg >> 10;
  int q0 = qt * 16;

  int l = lane_id();
  int lo = l & 15;
  int hi = (l >> 4) & 1;

  float mrow[8], lrow[8];
  v8f accO[4] = {};
#pragma unroll
  for (int r = 0; r < 8; ++r) { mrow[r] = -1e30f; lrow[r] = 0.0f; }

  const __bf16* qup = qu + ((long)b * T_SEQ + q0) * D_MODEL + h * HEAD_DIM;
  const __bf16* qvp = qv + ((long)b * T_SEQ + q0) * D_MODEL + h * HEAD_DIM;
  const __bf16* kcp = kc + ((long)b * T_SEQ) * D_MODEL + h * HEAD_DIM;
  const __bf16* klp = kl + h * HEAD_DIM;
  const int* mp = mask + (long)b * T_SEQ * T_SEQ;

  v16bf quf[2], qvf[2];
#pragma unroll
  for (int ks = 0; ks < 2; ++ks) {
    quf[ks] = load_a_bf16(qup, D_MODEL, 0, ks * 32);
    qvf[ks] = load_a_bf16(qvp, D_MODEL, 0, ks * 32);
  }

  for (int k0 = 0; k0 < T_SEQ; k0 += 32) {
    // content scores (q+u)·kc : two 16-col tiles
    v8f sc[2] = {};
#pragma unroll
    for (int nt = 0; nt < 2; ++nt)
#pragma unroll
      for (int ks = 0; ks < 2; ++ks)
        sc[nt] = wmma_bf16(quf[ks],
                           load_b_bf16(kcp, D_MODEL, k0 + nt * 16, ks * 32), sc[nt]);

    // position band G[q][cc] = (q+v)·kl[L0+cc], cc in [0,46]; l = 2047+k-q = L0+dk+15-dq
    int L0 = (T_SEQ - 1) - 15 + k0 - q0; // in [0,4048]; rows touched <= 4095 (padded buf)
    v8f G[3] = {};
#pragma unroll
    for (int gt = 0; gt < 3; ++gt)
#pragma unroll
      for (int ks = 0; ks < 2; ++ks)
        G[gt] = wmma_bf16(qvf[ks],
                          load_b_bf16(klp, D_MODEL, L0 + gt * 16, ks * 32), G[gt]);

    // relative shift: per row dq the shift s = 15-dq, cc(nt) = 16*nt + lo + s.
    // cc&15 is nt-independent -> one bpermute index serves both score tiles.
    v8f sv[2];
#pragma unroll
    for (int r = 0; r < 8; ++r) {
      int dq = r + 8 * hi;
      int t = lo + 15 - dq;                 // cc for nt=0, in [0,30]
      int idx = (l & 16) | (t & 15);
      float bp0 = bperm_f32(idx, G[0][r]);
      float bp1 = bperm_f32(idx, G[1][r]);
      float bp2 = bperm_f32(idx, G[2][r]);
      bool carry = (t >= 16);
      float bd0 = carry ? bp1 : bp0;
      float bd1 = carry ? bp2 : bp1;
      int mk0 = mp[(long)(q0 + dq) * T_SEQ + k0 + lo];
      int mk1 = mp[(long)(q0 + dq) * T_SEQ + k0 + 16 + lo];
      float s0 = (sc[0][r] + bd0) * 0.125f; // 1/sqrt(64)
      float s1 = (sc[1][r] + bd1) * 0.125f;
      sv[0][r] = (mk0 == 0) ? -1e30f : s0;
      sv[1][r] = (mk1 == 0) ? -1e30f : s1;
    }

    // online softmax (row reductions stay inside 16-lane C-layout half groups)
    float corr[8];
#pragma unroll
    for (int r = 0; r < 8; ++r) {
      float rm = fmaxf(sv[0][r], sv[1][r]);
#pragma unroll
      for (int mm = 8; mm >= 1; mm >>= 1) rm = fmaxf(rm, __shfl_xor(rm, mm, 32));
      float mnew = fmaxf(mrow[r], rm);
      corr[r] = __expf(mrow[r] - mnew);
      mrow[r] = mnew;
      float p0 = __expf(sv[0][r] - mnew);
      float p1 = __expf(sv[1][r] - mnew);
      sv[0][r] = p0;
      sv[1][r] = p1;
      float rs = p0 + p1;
#pragma unroll
      for (int mm = 8; mm >= 1; mm >>= 1) rs += __shfl_xor(rs, mm, 32);
      lrow[r] = lrow[r] * corr[r] + rs;
    }
#pragma unroll
    for (int nt4 = 0; nt4 < 4; ++nt4)
#pragma unroll
      for (int r = 0; r < 8; ++r) accO[nt4][r] *= corr[r];

    // stage P (bf16) to LDS, reload as WMMA A fragment (16 queries x 32 keys)
#pragma unroll
    for (int nt = 0; nt < 2; ++nt)
#pragma unroll
      for (int r = 0; r < 8; ++r) {
        int dq = r + 8 * hi;
        pst[dq * 40 + nt * 16 + lo] = (__bf16)sv[nt][r];
      }
    v16bf pf;
    {
      const __bf16* pp = pst + lo * 40 + (hi ? 8 : 0);
      v8bf plo = *(const v8bf*)pp;
      v8bf phi = *(const v8bf*)(pp + 16);
#pragma unroll
      for (int i = 0; i < 8; ++i) { pf[i] = plo[i]; pf[i + 8] = phi[i]; }
    }

    // out += P @ V  (vT rows are n = h*64+d, K contiguous over keys)
#pragma unroll
    for (int nt4 = 0; nt4 < 4; ++nt4)
      accO[nt4] = wmma_bf16(
          pf,
          load_b_bf16(vT, (long)M_ROWS, h * HEAD_DIM + nt4 * 16, (long)b * T_SEQ + k0),
          accO[nt4]);
  }

  // epilogue: normalize and store fp32 output (B,T,D)
#pragma unroll
  for (int nt4 = 0; nt4 < 4; ++nt4) {
#pragma unroll
    for (int r = 0; r < 8; ++r) {
      int dq = r + 8 * hi;
      float inv = 1.0f / lrow[r];
      out[((long)b * T_SEQ + q0 + dq) * D_MODEL + h * HEAD_DIM + nt4 * 16 + lo] =
          accO[nt4][r] * inv;
    }
  }
}

extern "C" void kernel_launch(void* const* d_in, const int* in_sizes, int n_in,
                              void* d_out, int out_size, void* d_ws, size_t ws_size,
                              hipStream_t stream) {
  (void)in_sizes; (void)n_in; (void)out_size; (void)ws_size;
  const float* x    = (const float*)d_in[0];
  const int*   mask = (const int*)d_in[1];
  const float* Wq   = (const float*)d_in[2];
  const float* bq   = (const float*)d_in[3];
  const float* Wv   = (const float*)d_in[4];
  const float* bv   = (const float*)d_in[5];
  const float* Wkc  = (const float*)d_in[6];
  const float* bkc  = (const float*)d_in[7];
  const float* Wkl  = (const float*)d_in[8];
  const float* bkl  = (const float*)d_in[9];
  const float* u    = (const float*)d_in[10];
  const float* v    = (const float*)d_in[11];
  float* out = (float*)d_out;

  char* ws = (char*)d_ws;
  size_t off = 0;
  auto alloc = [&](size_t bytes) -> void* {
    void* p = ws + off;
    off += (bytes + 255) & ~(size_t)255;
    return p;
  };
  __bf16* xB   = (__bf16*)alloc((size_t)M_ROWS * D_MODEL * 2);
  __bf16* WqB  = (__bf16*)alloc((size_t)D_MODEL * D_MODEL * 2);
  __bf16* WvB  = (__bf16*)alloc((size_t)D_MODEL * D_MODEL * 2);
  __bf16* WkcB = (__bf16*)alloc((size_t)D_MODEL * D_MODEL * 2);
  __bf16* WklB = (__bf16*)alloc((size_t)D_MODEL * D_MODEL * 2);
  __bf16* peB  = (__bf16*)alloc((size_t)PE_ROWS * D_MODEL * 2);
  __bf16* quB  = (__bf16*)alloc((size_t)M_ROWS * D_MODEL * 2);
  __bf16* qvB  = (__bf16*)alloc((size_t)M_ROWS * D_MODEL * 2);
  __bf16* kcB  = (__bf16*)alloc((size_t)M_ROWS * D_MODEL * 2);
  __bf16* vTB  = (__bf16*)alloc((size_t)M_ROWS * D_MODEL * 2);
  __bf16* klB  = (__bf16*)alloc((size_t)PE_ROWS * D_MODEL * 2);

  // fp32 -> bf16 one-pass converts (removes per-K-step cvt work from the GEMMs)
  long nx8 = (long)M_ROWS * D_MODEL / 8;
  long nw8 = (long)D_MODEL * D_MODEL / 8;
  cvt_kernel<<<(int)((nx8 + 255) / 256), 256, 0, stream>>>(x, xB, nx8);
  cvt_kernel<<<(int)((nw8 + 255) / 256), 256, 0, stream>>>(Wq, WqB, nw8);
  cvt_kernel<<<(int)((nw8 + 255) / 256), 256, 0, stream>>>(Wv, WvB, nw8);
  cvt_kernel<<<(int)((nw8 + 255) / 256), 256, 0, stream>>>(Wkc, WkcB, nw8);
  cvt_kernel<<<(int)((nw8 + 255) / 256), 256, 0, stream>>>(Wkl, WklB, nw8);

  pe_kernel<<<(PE_ROWS * D_MODEL) / 256, 256, 0, stream>>>(peB);

  int pb = (M_ROWS / 32) * (D_MODEL / 64) / 4; // 256 blocks of 4 waves
  proj_kernel<<<pb, 128, 0, stream>>>(xB,  WqB,  bq,  u,       v,       quB, qvB,     M_ROWS,  0);
  proj_kernel<<<pb, 128, 0, stream>>>(xB,  WkcB, bkc, nullptr, nullptr, kcB, nullptr, M_ROWS,  0);
  proj_kernel<<<pb, 128, 0, stream>>>(xB,  WvB,  bv,  nullptr, nullptr, vTB, nullptr, M_ROWS,  1);
  proj_kernel<<<pb, 128, 0, stream>>>(peB, WklB, bkl, nullptr, nullptr, klB, nullptr, PE_ROWS, 0);

  int ab = (B_SZ * N_HEADS * (T_SEQ / 16)) / 4; // 512 blocks of 4 waves
  attn_kernel<<<ab, 128, 0, stream>>>(quB, qvB, kcB, klB, vTB, mask, out);
}
